// TransformerBlock_63668595196337
// MI455X (gfx1250) — matrix operand order repaired and checked by reading, not compiled, not run on hardware
//
#include <hip/hip_runtime.h>
#include <hip/hip_bf16.h>

// ---------------------------------------------------------------------------
// Types for WMMA fragments (gfx1250, wave32)
// ---------------------------------------------------------------------------
typedef __attribute__((ext_vector_type(16))) __bf16 v16bf;
typedef __attribute__((ext_vector_type(8)))  __bf16 v8bf;
typedef __attribute__((ext_vector_type(8)))  float  v8f;

static constexpr int Bv  = 2;
static constexpr int Tv  = 2048;
static constexpr int Dv  = 1024;
static constexpr int Hv  = 16;
static constexpr int HDv = 64;
static constexpr int FFv = 4096;
static constexpr float EPSv = 1e-6f;

__device__ __forceinline__ __bf16 f2bf(float f) {
    unsigned int u = __builtin_bit_cast(unsigned int, f);
    unsigned int r = (u + 0x7fffu + ((u >> 16) & 1u)) >> 16;
    unsigned short s = (unsigned short)r;
    return __builtin_bit_cast(__bf16, s);
}
__device__ __forceinline__ unsigned short bfbits(float f) {
    return __builtin_bit_cast(unsigned short, f2bf(f));
}
__device__ __forceinline__ v16bf cat16(v8bf lo, v8bf hi) {
    return __builtin_shufflevector(lo, hi, 0,1,2,3,4,5,6,7,8,9,10,11,12,13,14,15);
}

// gfx1250 async global->LDS copy (16 bytes / lane), tracked by ASYNCcnt.
// VDST = per-lane LDS byte address (low 32 bits of the flat LDS pointer),
// VADDR = per-lane 64-bit global address.
__device__ __forceinline__ void async_copy_b128(const void* gptr, void* lptr) {
    unsigned int      l = (unsigned int)(unsigned long long)lptr;
    unsigned long long g = (unsigned long long)gptr;
    asm volatile("global_load_async_to_lds_b128 %0, %1, off"
                 :: "v"(l), "v"(g) : "memory");
}
__device__ __forceinline__ void wait_async0() {
    asm volatile("s_wait_asynccnt 0x0" ::: "memory");
}

// ---------------------------------------------------------------------------
// fp32 -> bf16 conversion (vectorized: float4 in, ushort4 out)
// ---------------------------------------------------------------------------
__global__ void k_cvt4(const float4* __restrict__ in, ushort4* __restrict__ out, int n4) {
    int i = blockIdx.x * blockDim.x + threadIdx.x;
    if (i < n4) {
        float4 f = in[i];
        ushort4 o;
        o.x = bfbits(f.x); o.y = bfbits(f.y); o.z = bfbits(f.z); o.w = bfbits(f.w);
        out[i] = o;
    }
}

// ---------------------------------------------------------------------------
// RMSNorm kernels
// ---------------------------------------------------------------------------
__global__ __launch_bounds__(256)
void k_rmsnorm(const float* __restrict__ x, const float* __restrict__ w,
               float* __restrict__ of, __bf16* __restrict__ ob) {
    __shared__ float red[256];
    int row = blockIdx.x, tid = threadIdx.x;
    const float* xr = x + (size_t)row * Dv;
    float vals[4]; float ss = 0.f;
#pragma unroll
    for (int i = 0; i < 4; ++i) { vals[i] = xr[tid + i * 256]; ss += vals[i] * vals[i]; }
    red[tid] = ss; __syncthreads();
    for (int s = 128; s > 0; s >>= 1) { if (tid < s) red[tid] += red[tid + s]; __syncthreads(); }
    float rms = rsqrtf(red[0] * (1.0f / Dv) + EPSv);
#pragma unroll
    for (int i = 0; i < 4; ++i) {
        int c = tid + i * 256;
        float v = vals[i] * rms * w[c];
        of[(size_t)row * Dv + c] = v;
        if (ob) ob[(size_t)row * Dv + c] = f2bf(v);
    }
}

__global__ __launch_bounds__(256)
void k_add_rmsnorm(const float* __restrict__ a, const float* __restrict__ bsrc,
                   const float* __restrict__ w,
                   float* __restrict__ of, __bf16* __restrict__ ob) {
    __shared__ float red[256];
    int row = blockIdx.x, tid = threadIdx.x;
    const float* ar = a    + (size_t)row * Dv;
    const float* br = bsrc + (size_t)row * Dv;
    float vals[4]; float ss = 0.f;
#pragma unroll
    for (int i = 0; i < 4; ++i) {
        int c = tid + i * 256;
        vals[i] = ar[c] + br[c];
        ss += vals[i] * vals[i];
    }
    red[tid] = ss; __syncthreads();
    for (int s = 128; s > 0; s >>= 1) { if (tid < s) red[tid] += red[tid + s]; __syncthreads(); }
    float rms = rsqrtf(red[0] * (1.0f / Dv) + EPSv);
#pragma unroll
    for (int i = 0; i < 4; ++i) {
        int c = tid + i * 256;
        float v = vals[i] * rms * w[c];
        of[(size_t)row * Dv + c] = v;
        if (ob) ob[(size_t)row * Dv + c] = f2bf(v);
    }
}

// ---------------------------------------------------------------------------
// bf16 WMMA GEMM: C[M,N] = A[M,K] * Bw[K,N]  (row major, f32 accumulate)
// Block 256 thr = 8 waves (4x2), block tile 64x128, wave tile 16x64, K-step 32.
// Double-buffered LDS; A staged via global_load_async_to_lds_b128 (ASYNCcnt),
// B staged via registers with an in-flight transpose (BsT[col][k]) so B
// fragments are two ds_load_b128. One barrier per K-step.
// ---------------------------------------------------------------------------
template<int ACT, int OBF>
__global__ __launch_bounds__(256)
void k_gemm(const __bf16* __restrict__ A, const __bf16* __restrict__ Bw,
            float* __restrict__ Cf, __bf16* __restrict__ Cb,
            int M, int N, int K) {
    __shared__ __bf16 As[2][64][40];    // [buf][row][k]
    __shared__ __bf16 BsT[2][128][40];  // [buf][col][k]
    (void)M;
    const int tid  = threadIdx.x;
    const int wave = tid >> 5, lane = tid & 31;
    const int hv   = lane >> 4, lm = lane & 15;
    const int wrow = wave >> 1, wcol = wave & 1;   // 4x2 wave grid
    const int rowBase = blockIdx.y * 64;
    const int colBase = blockIdx.x * 128;

    // staging assignments: A = 8 elems (one b128) / thread, B = 16 elems / thread
    const int ar = tid >> 2, ac = (tid & 3) * 8;
    const int br = tid >> 3, bc = (tid & 7) * 16;

    const __bf16* gaBase = A  + (size_t)(rowBase + ar) * K + ac;
    const __bf16* gbBase = Bw + (size_t)br * N + colBase + bc;

    v8f acc[4] = { {}, {}, {}, {} };
    const int nIter = K / 32;

    // ---- prologue: stage tile 0 into buffer 0 ----
    async_copy_b128(gaBase, &As[0][ar][ac]);
    {
        const __bf16* gb = gbBase;
        ushort4 b0 = ((const ushort4*)gb)[0];
        ushort4 b1 = ((const ushort4*)gb)[1];
        ushort4 b2 = ((const ushort4*)gb)[2];
        ushort4 b3 = ((const ushort4*)gb)[3];
        unsigned short bsv[16] = { b0.x,b0.y,b0.z,b0.w, b1.x,b1.y,b1.z,b1.w,
                                   b2.x,b2.y,b2.z,b2.w, b3.x,b3.y,b3.z,b3.w };
#pragma unroll
        for (int j = 0; j < 16; ++j)
            BsT[0][bc + j][br] = __builtin_bit_cast(__bf16, bsv[j]);
    }
    wait_async0();
    __syncthreads();

    for (int i = 0; i < nIter; ++i) {
        const int cur = i & 1, nxt = cur ^ 1;
        const bool hasNext = (i + 1) < nIter;

        // ---- issue next tile: async A -> LDS, B -> registers ----
        ushort4 nb0 = {}, nb1 = {}, nb2 = {}, nb3 = {};
        const __bf16* gbN = gbBase + (size_t)(i + 1) * 32 * N;
        if (hasNext) {
            async_copy_b128(gaBase + (i + 1) * 32, &As[nxt][ar][ac]);
            nb0 = ((const ushort4*)gbN)[0];
            nb1 = ((const ushort4*)gbN)[1];
            nb2 = ((const ushort4*)gbN)[2];
            nb3 = ((const ushort4*)gbN)[3];
            if (i + 2 < nIter) __builtin_prefetch(gbN + (size_t)32 * N, 0, 1);
        }

        // ---- compute on current buffer ----
        const int arow = wrow * 16 + lm;
        v8bf alo = *(const v8bf*)&As[cur][arow][hv * 8];
        v8bf ahi = *(const v8bf*)&As[cur][arow][16 + hv * 8];
        v16bf afrag = cat16(alo, ahi);
#pragma unroll
        for (int n = 0; n < 4; ++n) {
            const int bcol = wcol * 64 + n * 16 + lm;
            v8bf blo = *(const v8bf*)&BsT[cur][bcol][hv * 8];
            v8bf bhi = *(const v8bf*)&BsT[cur][bcol][16 + hv * 8];
            v16bf bfrag = cat16(blo, bhi);
            acc[n] = __builtin_amdgcn_wmma_f32_16x16x32_bf16(
                         false, afrag, false, bfrag, (short)0, acc[n], false, false);
        }

        // ---- commit next B tile (transposed) and close the pipeline stage ----
        if (hasNext) {
            unsigned short bsv[16] = { nb0.x,nb0.y,nb0.z,nb0.w, nb1.x,nb1.y,nb1.z,nb1.w,
                                       nb2.x,nb2.y,nb2.z,nb2.w, nb3.x,nb3.y,nb3.z,nb3.w };
#pragma unroll
            for (int j = 0; j < 16; ++j)
                BsT[nxt][bc + j][br] = __builtin_bit_cast(__bf16, bsv[j]);
            wait_async0();
        }
        __syncthreads();
    }

    // epilogue: acc[n][r] -> row = wrow*16 + r + hv*8 ; col = wcol*64 + n*16 + lm
#pragma unroll
    for (int n = 0; n < 4; ++n) {
        const int col = colBase + wcol * 64 + n * 16 + lm;
#pragma unroll
        for (int r = 0; r < 8; ++r) {
            const int row = rowBase + wrow * 16 + r + hv * 8;
            float v = acc[n][r];
            if (ACT == 1) v = v / (1.0f + __expf(-v));   // SiLU
            size_t off = (size_t)row * N + col;
            if (OBF) Cb[off] = f2bf(v);
            else     Cf[off] = v;
        }
    }
}

// ---------------------------------------------------------------------------
// Flash attention. Block = 128 threads = 4 waves; each wave owns one 16-query
// tile; K (async, row-major) and V (transposed) staged cooperatively per
// 32-key block. Online softmax, causal. Layout [B,T,D], head h at cols h*64.
// ---------------------------------------------------------------------------
__global__ __launch_bounds__(128)
void k_attn(const __bf16* __restrict__ Qb, const __bf16* __restrict__ Kb,
            const __bf16* __restrict__ Vb, __bf16* __restrict__ Ob) {
    __shared__ __bf16 Qs[4][16][80];  // per-wave Q tile [q][k]
    __shared__ __bf16 Ks[32][80];     // K block [key][k]
    __shared__ __bf16 VsT[64][40];    // V block transposed [dim][key]
    __shared__ __bf16 Ps[4][16][40];  // per-wave P transpose scratch [q][key]

    const int tid  = threadIdx.x;
    const int wave = tid >> 5, lane = tid & 31;
    const int hv   = lane >> 4, lm = lane & 15;
    const int q0   = (blockIdx.x * 4 + wave) * 16;
    const int bh   = blockIdx.y;
    const int b    = bh >> 4, h = bh & 15;
    const size_t base = (size_t)b * Tv * Dv + (size_t)h * HDv;

    // ---- stage own Q tile 16x64 via async copies ----
#pragma unroll
    for (int it = 0; it < 4; ++it) {
        int idx8 = it * 32 + lane;                 // 128 x 16-byte chunks
        int r = idx8 >> 3, c8 = (idx8 & 7) * 8;
        async_copy_b128(Qb + base + (size_t)(q0 + r) * Dv + c8, &Qs[wave][r][c8]);
    }
    wait_async0();
    __syncthreads();

    v16bf aq[2];
#pragma unroll
    for (int s = 0; s < 2; ++s) {
        v8bf lo = *(const v8bf*)&Qs[wave][lm][s * 32 + hv * 8];
        v8bf hi = *(const v8bf*)&Qs[wave][lm][s * 32 + 16 + hv * 8];
        aq[s] = cat16(lo, hi);
    }

    float m[8], l[8];
#pragma unroll
    for (int r = 0; r < 8; ++r) { m[r] = -1e30f; l[r] = 0.f; }
    v8f Oacc[4] = { {}, {}, {}, {} };

    const int kbmax_w   = __builtin_amdgcn_readfirstlane((q0 + 15) >> 5); // wave-uniform (scalar)
    const int kbmax_blk = (blockIdx.x * 64 + 63) >> 5;

    for (int kb = 0; kb <= kbmax_blk; ++kb) {
        __syncthreads();
        // ---- cooperative staging: K via async (row-major), V transposed ----
#pragma unroll
        for (int it = 0; it < 2; ++it) {
            int idx8 = it * 128 + tid;             // 256 x 16-byte chunks
            int r = idx8 >> 3, c8 = (idx8 & 7) * 8;
            async_copy_b128(Kb + base + (size_t)(kb * 32 + r) * Dv + c8, &Ks[r][c8]);
        }
#pragma unroll
        for (int it = 0; it < 4; ++it) {
            int idx4 = it * 128 + tid;
            int r = idx4 >> 4, c4 = idx4 & 15;
            ushort4 vv = *(const ushort4*)(Vb + base + (size_t)(kb * 32 + r) * Dv + c4 * 4);
            VsT[c4 * 4 + 0][r] = __builtin_bit_cast(__bf16, vv.x);
            VsT[c4 * 4 + 1][r] = __builtin_bit_cast(__bf16, vv.y);
            VsT[c4 * 4 + 2][r] = __builtin_bit_cast(__bf16, vv.z);
            VsT[c4 * 4 + 3][r] = __builtin_bit_cast(__bf16, vv.w);
        }
        wait_async0();
        __syncthreads();

        if (kb <= kbmax_w) {   // scalarized wave-uniform branch: EXEC stays all-ones
            // ---- S[t] = Q . K_sub^T ----
            v8f S[2] = { {}, {} };
#pragma unroll
            for (int t = 0; t < 2; ++t) {
#pragma unroll
                for (int s = 0; s < 2; ++s) {
                    v8bf klo = *(const v8bf*)&Ks[t * 16 + lm][s * 32 + hv * 8];
                    v8bf khi = *(const v8bf*)&Ks[t * 16 + lm][s * 32 + 16 + hv * 8];
                    v16bf bk = cat16(klo, khi);
                    S[t] = __builtin_amdgcn_wmma_f32_16x16x32_bf16(
                               false, aq[s], false, bk, (short)0, S[t], false, false);
                }
            }

            // ---- scale + causal mask ----
#pragma unroll
            for (int t = 0; t < 2; ++t) {
                int key = kb * 32 + t * 16 + lm;
#pragma unroll
                for (int r = 0; r < 8; ++r) {
                    int qrow = q0 + r + hv * 8;
                    float v = S[t][r] * 0.125f;            // 1/sqrt(64)
                    S[t][r] = (key > qrow) ? -1e30f : v;
                }
            }

            // ---- online softmax ----
#pragma unroll
            for (int r = 0; r < 8; ++r) {
                float mx = fmaxf(S[0][r], S[1][r]);
                mx = fmaxf(mx, __shfl_xor(mx, 1, 32));
                mx = fmaxf(mx, __shfl_xor(mx, 2, 32));
                mx = fmaxf(mx, __shfl_xor(mx, 4, 32));
                mx = fmaxf(mx, __shfl_xor(mx, 8, 32));
                float mnew = fmaxf(m[r], mx);
                float corr = __expf(m[r] - mnew);
                float p0 = __expf(S[0][r] - mnew);
                float p1 = __expf(S[1][r] - mnew);
                float rs = p0 + p1;
                rs += __shfl_xor(rs, 1, 32);
                rs += __shfl_xor(rs, 2, 32);
                rs += __shfl_xor(rs, 4, 32);
                rs += __shfl_xor(rs, 8, 32);
                l[r] = l[r] * corr + rs;
                m[r] = mnew;
                S[0][r] = p0; S[1][r] = p1;
                Oacc[0][r] *= corr; Oacc[1][r] *= corr;
                Oacc[2][r] *= corr; Oacc[3][r] *= corr;
            }

            // ---- P: C layout -> A layout via per-wave LDS scratch ----
#pragma unroll
            for (int r = 0; r < 8; ++r) {
                Ps[wave][r + hv * 8][lm]      = f2bf(S[0][r]);
                Ps[wave][r + hv * 8][16 + lm] = f2bf(S[1][r]);
            }
            asm volatile("s_wait_dscnt 0x0" ::: "memory");  // wave-private ordering
            v8bf plo = *(const v8bf*)&Ps[wave][lm][hv * 8];
            v8bf phi = *(const v8bf*)&Ps[wave][lm][16 + hv * 8];
            v16bf ap = cat16(plo, phi);

            // ---- O += P(16x32) . V(32x64) ----
#pragma unroll
            for (int sub = 0; sub < 4; ++sub) {
                v8bf vlo = *(const v8bf*)&VsT[sub * 16 + lm][hv * 8];
                v8bf vhi = *(const v8bf*)&VsT[sub * 16 + lm][16 + hv * 8];
                v16bf bv = cat16(vlo, vhi);
                Oacc[sub] = __builtin_amdgcn_wmma_f32_16x16x32_bf16(
                                false, ap, false, bv, (short)0, Oacc[sub], false, false);
            }
        }
    }

    // ---- normalize, write back bf16 [B,T,D] ----
#pragma unroll
    for (int r = 0; r < 8; ++r) {
        float inv = 1.0f / l[r];
        int row = q0 + r + hv * 8;
        size_t o = base + (size_t)row * Dv;
        Ob[o +  0 + lm] = f2bf(Oacc[0][r] * inv);
        Ob[o + 16 + lm] = f2bf(Oacc[1][r] * inv);
        Ob[o + 32 + lm] = f2bf(Oacc[2][r] * inv);
        Ob[o + 48 + lm] = f2bf(Oacc[3][r] * inv);
    }
}

// ---------------------------------------------------------------------------
// Host-side orchestration
// ---------------------------------------------------------------------------
extern "C" void kernel_launch(void* const* d_in, const int* in_sizes, int n_in,
                              void* d_out, int out_size, void* d_ws, size_t ws_size,
                              hipStream_t stream) {
    (void)in_sizes; (void)n_in; (void)out_size; (void)ws_size;
    const float* x    = (const float*)d_in[0];
    const float* wpre = (const float*)d_in[1];
    const float* wq   = (const float*)d_in[2];
    const float* wk   = (const float*)d_in[3];
    const float* wv   = (const float*)d_in[4];
    const float* wo   = (const float*)d_in[5];
    const float* wan  = (const float*)d_in[6];
    const float* w1   = (const float*)d_in[7];
    const float* w2   = (const float*)d_in[8];
    const float* wfn  = (const float*)d_in[9];

    const size_t MT = (size_t)Bv * Tv;   // 4096 rows
    char*  ws  = (char*)d_ws;
    size_t off = 0;
    auto alloc = [&](size_t bytes) -> char* {
        char* p = ws + off;
        off += (bytes + 255) & ~(size_t)255;
        return p;
    };

    float*  h_f  = (float*) alloc(MT * Dv * 4);
    __bf16* h_b  = (__bf16*)alloc(MT * Dv * 2);
    __bf16* qb   = (__bf16*)alloc(MT * Dv * 2);
    __bf16* kb_  = (__bf16*)alloc(MT * Dv * 2);
    __bf16* vb   = (__bf16*)alloc(MT * Dv * 2);
    __bf16* ab   = (__bf16*)alloc(MT * Dv * 2);
    float*  proj = (float*) alloc(MT * Dv * 4);
    float*  y_f  = (float*) alloc(MT * Dv * 4);
    __bf16* y_b  = (__bf16*)alloc(MT * Dv * 2);
    __bf16* mid  = (__bf16*)alloc(MT * FFv * 2);
    float*  ffn  = (float*) alloc(MT * Dv * 4);
    __bf16* wqb  = (__bf16*)alloc((size_t)Dv * Dv * 2);
    __bf16* wkb  = (__bf16*)alloc((size_t)Dv * Dv * 2);
    __bf16* wvb  = (__bf16*)alloc((size_t)Dv * Dv * 2);
    __bf16* wob  = (__bf16*)alloc((size_t)Dv * Dv * 2);
    __bf16* w1b  = (__bf16*)alloc((size_t)Dv * FFv * 2);
    __bf16* w2b  = (__bf16*)alloc((size_t)Dv * FFv * 2);

    // weights -> bf16 (vectorized)
    {
        int nD4 = Dv * Dv / 4, nF4 = Dv * FFv / 4;
        k_cvt4<<<(nD4 + 255) / 256, 256, 0, stream>>>((const float4*)wq, (ushort4*)wqb, nD4);
        k_cvt4<<<(nD4 + 255) / 256, 256, 0, stream>>>((const float4*)wk, (ushort4*)wkb, nD4);
        k_cvt4<<<(nD4 + 255) / 256, 256, 0, stream>>>((const float4*)wv, (ushort4*)wvb, nD4);
        k_cvt4<<<(nD4 + 255) / 256, 256, 0, stream>>>((const float4*)wo, (ushort4*)wob, nD4);
        k_cvt4<<<(nF4 + 255) / 256, 256, 0, stream>>>((const float4*)w1, (ushort4*)w1b, nF4);
        k_cvt4<<<(nF4 + 255) / 256, 256, 0, stream>>>((const float4*)w2, (ushort4*)w2b, nF4);
    }

    // pre-norm
    k_rmsnorm<<<(int)MT, 256, 0, stream>>>(x, wpre, h_f, h_b);

    // Q/K/V projections (bf16 out); block tile 64x128
    dim3 gD(Dv / 128, (int)MT / 64), blk(256);
    k_gemm<0, 1><<<gD, blk, 0, stream>>>(h_b, wqb, nullptr, qb,  (int)MT, Dv, Dv);
    k_gemm<0, 1><<<gD, blk, 0, stream>>>(h_b, wkb, nullptr, kb_, (int)MT, Dv, Dv);
    k_gemm<0, 1><<<gD, blk, 0, stream>>>(h_b, wvb, nullptr, vb,  (int)MT, Dv, Dv);

    // causal flash attention (4 q-tiles per block)
    k_attn<<<dim3(Tv / 64, Bv * Hv), 128, 0, stream>>>(qb, kb_, vb, ab);

    // output projection (f32 out) + residual + norm -> y
    k_gemm<0, 0><<<gD, blk, 0, stream>>>(ab, wob, proj, nullptr, (int)MT, Dv, Dv);
    k_add_rmsnorm<<<(int)MT, 256, 0, stream>>>(h_f, proj, wan, y_f, y_b);

    // FFN: silu(y @ W1) @ W2 ; residual + norm -> out
    dim3 gF(FFv / 128, (int)MT / 64);
    k_gemm<1, 1><<<gF, blk, 0, stream>>>(y_b, w1b, nullptr, mid, (int)MT, FFv, Dv);
    k_gemm<0, 0><<<gD, blk, 0, stream>>>(mid, w2b, ffn, nullptr, (int)MT, Dv, FFv);
    k_add_rmsnorm<<<(int)MT, 256, 0, stream>>>(y_f, ffn, wfn, (float*)d_out, nullptr);
}
